// AnisotropicGaussianSampler_49916109914679
// MI455X (gfx1250) — matrix operand order repaired and checked by reading, not compiled, not run on hardware
//
#include <hip/hip_runtime.h>
#include <hip/hip_bf16.h>

typedef _Float16 v16h __attribute__((ext_vector_type(16)));
typedef _Float16 v8h  __attribute__((ext_vector_type(8)));
typedef float    v8f  __attribute__((ext_vector_type(8)));

#define HH   128
#define WW   128
#define NTOT 4096
#define BB   8

union V16 { v16h v; v8h h[2]; };

// ---- Kernel 1: convert f32 activations -> f16 (staged once in d_ws) ----
__global__ void cvt_f32_to_f16(const float* __restrict__ src,
                               _Float16* __restrict__ dst, int n) {
    int i = blockIdx.x * blockDim.x + threadIdx.x;
    if (i < n) dst[i] = (_Float16)src[i];
}

// ---- Kernel 2: fused separable-Gaussian contraction via WMMA ----
// Each block: one 16-wide n-tile. 8 waves, wave w handles batch b=w.
__global__ __launch_bounds__(256)
void gauss_wmma_kernel(const _Float16* __restrict__ Ah,   // [8][128][128] f16
                       const float* __restrict__ mu,      // [4096][2]
                       const float* __restrict__ sigma,   // [4096][2]
                       float* __restrict__ out)           // [8][4096]
{
    __shared__ __align__(16) _Float16 ewT[16 * WW];  // [n][w]  (B operand, transposed)
    __shared__ float                  ehs[16 * HH];  // [n][h]  (reduction weights, f32)

    const int tid   = threadIdx.x;
    const int lane  = tid & 31;
    const int nbase = blockIdx.x * 16;

    // --- cooperative fill of Ew (f16, transposed) and Eh (f32) ---
    for (int idx = tid; idx < 16 * 128; idx += 256) {
        const int n  = idx >> 7;     // 0..15 within tile
        const int x  = idx & 127;    // spatial coordinate (h or w)
        const int gn = nbase + n;
        const float muh = mu[gn * 2 + 0];
        const float muw = mu[gn * 2 + 1];
        const float sh  = sigma[gn * 2 + 0];
        const float sw  = sigma[gn * 2 + 1];
        const float ch  = 1.0f / (2.0f * sh * sh);
        const float cw  = 1.0f / (2.0f * sw * sw);
        const float g   = (float)x * (1.0f / 128.0f);
        const float dh  = g - muh;
        const float dw  = g - muw;
        ehs[idx] = __expf(-dh * dh * ch);
        ewT[idx] = (_Float16)__expf(-dw * dw * cw);
    }
    __syncthreads();

    const int b   = tid >> 5;        // wave id == batch index
    const int rlo = lane & 15;       // A: row-in-half-tile; B/D: column n
    const int hi  = lane >> 4;       // which K/M half this lane covers

    const _Float16* __restrict__ Ab = Ah + b * (HH * WW);

    float acc = 0.0f;
    for (int ht = 0; ht < 8; ++ht) {              // 8 h-tiles of 16 rows
        v8f c = {};
        const _Float16* __restrict__ Arow = Ab + (ht * 16 + rlo) * WW;
#pragma unroll
        for (int kc = 0; kc < 4; ++kc) {          // K = 128 in 4 steps of 32
            V16 a, bmat;
            const int k0 = kc * 32 + hi * 8;
            a.h[0] = *(const v8h*)(Arow + k0);            // K[k0 .. k0+7]
            a.h[1] = *(const v8h*)(Arow + k0 + 16);       // K[k0+16 .. k0+23]
            const _Float16* brow = &ewT[rlo * WW + kc * 32 + hi * 8];
            bmat.h[0] = *(const v8h*)(brow);              // K chunk, col = rlo
            bmat.h[1] = *(const v8h*)(brow + 16);
            c = __builtin_amdgcn_wmma_f32_16x16x32_f16(
                    /*neg_a=*/false, a.v, /*neg_b=*/false, bmat.v,
                    /*c_mod=*/(short)0, c, /*reuse_a=*/false, /*reuse_b=*/false);
        }
        // weighted reduction over the 16 rows of this h-tile:
        // VGPR r holds T[m, n] with m = ht*16 + r + 8*hi, n = rlo
#pragma unroll
        for (int r = 0; r < 8; ++r) {
            const int m = ht * 16 + r + hi * 8;
            acc += ehs[rlo * HH + m] * c[r];
        }
    }

    // combine the two M-halves living in lanes l and l^16 (wave32)
    acc += __shfl_xor(acc, 16, 32);
    if (lane < 16) {
        out[b * NTOT + nbase + rlo] = acc * (1.0f / 16384.0f);
    }
}

extern "C" void kernel_launch(void* const* d_in, const int* in_sizes, int n_in,
                              void* d_out, int out_size, void* d_ws, size_t ws_size,
                              hipStream_t stream) {
    const float* act   = (const float*)d_in[0];   // [8,128,128] f32
    const float* mu    = (const float*)d_in[1];   // [4096,2]    f32
    const float* sigma = (const float*)d_in[2];   // [4096,2]    f32
    float*       out   = (float*)d_out;           // [8,64,64] -> flat [8,4096]

    _Float16* Ah = (_Float16*)d_ws;               // 131072 halves = 256 KB scratch
    const int nA = in_sizes[0];                   // 8*128*128 = 131072

    cvt_f32_to_f16<<<(nA + 255) / 256, 256, 0, stream>>>(act, Ah, nA);
    gauss_wmma_kernel<<<NTOT / 16, 256, 0, stream>>>(Ah, mu, sigma, out);
}
